// BaseModel_17867063952120
// MI455X (gfx1250) — matrix operand order repaired and checked by reading, not compiled, not run on hardware
//
#include <hip/hip_runtime.h>

typedef _Float16     v16h __attribute__((ext_vector_type(16)));
typedef _Float16     v2h  __attribute__((ext_vector_type(2)));
typedef float        v8f  __attribute__((ext_vector_type(8)));
typedef unsigned int v4u  __attribute__((ext_vector_type(4)));

static __device__ __forceinline__ v8f wmma16(v16h a, v16h b, v8f c) {
  // D = A(16x32 f16) * B(32x16 f16) + C(16x16 f32)
  return __builtin_amdgcn_wmma_f32_16x16x32_f16(false, a, false, b, (short)0, c,
                                                false, false);
}

static __device__ __forceinline__ v16h join16(v4u lo, v4u hi) {
  union { v4u q[2]; v16h h; } u;
  u.q[0] = lo; u.q[1] = hi;
  return u.h;
}

static __device__ __forceinline__ float leaky(float v) {
  return v > 0.f ? v : 0.01f * v;
}

// Grid: (BATCH/128, 128 t's). Block: 128 threads = 4 waves.
// Wave w handles batch rows [blk*128 + w*32, +32) for node t = blockIdx.y.
__global__ __launch_bounds__(128) void causal_mlp_kernel(
    const float* __restrict__ x,                       // [B,128]
    const float* __restrict__ w0g, const float* __restrict__ b0g,  // [128,64,128],[128,64]
    const float* __restrict__ w1g, const float* __restrict__ b1g,  // [128,64,64],[128,64]
    const float* __restrict__ w2g, const float* __restrict__ b2g,  // [128,2,64],[128,2]
    float* __restrict__ out)                           // [B,128,2]
{
  // padded strides (halves) to avoid LDS bank conflicts on b128 fragment loads
  __shared__ __align__(16) _Float16 s_w0[64 * 136];   // w0[t][i][j]  (i row, j col)
  __shared__ __align__(16) _Float16 s_w1[64 * 72];    // w1[t][i][j]
  __shared__ __align__(16) _Float16 s_h [128 * 72];   // h / h2 [local row][i]
  __shared__ float s_b0[64], s_b1[64], s_diag[64], s_xcol[128], s_w2[128], s_b2[2];

  const int  tid  = threadIdx.x;
  const int  t    = blockIdx.y;
  const long row0 = (long)blockIdx.x * 128;

  // ---- stage w0[t]: 64x128 f32 -> f16, row stride 136 halves
  {
    const float2* g = (const float2*)(w0g + (size_t)t * 64 * 128);
    #pragma unroll
    for (int it = 0; it < 32; ++it) {
      int idx = it * 128 + tid;          // 0..4095 float2's
      int r = idx >> 6, c = idx & 63;
      float2 v = g[idx];
      v2h h; h[0] = (_Float16)v.x; h[1] = (_Float16)v.y;
      *(v2h*)&s_w0[r * 136 + 2 * c] = h;
    }
  }
  // ---- stage w1[t]: 64x64 f32 -> f16, row stride 72 halves
  {
    const float2* g = (const float2*)(w1g + (size_t)t * 64 * 64);
    #pragma unroll
    for (int it = 0; it < 16; ++it) {
      int idx = it * 128 + tid;          // 0..2047 float2's
      int r = idx >> 5, c = idx & 31;
      float2 v = g[idx];
      v2h h; h[0] = (_Float16)v.x; h[1] = (_Float16)v.y;
      *(v2h*)&s_w1[r * 72 + 2 * c] = h;
    }
  }
  // ---- small stages (f32)
  if (tid < 64) {
    s_b0[tid]   = b0g[t * 64 + tid];
    s_b1[tid]   = b1g[t * 64 + tid];
    s_diag[tid] = w0g[(size_t)t * 8192 + (size_t)tid * 128 + t]; // w0[t,i,t]
  }
  s_xcol[tid] = x[(size_t)(row0 + tid) * 128 + t];               // x[b, t]
  s_w2[tid]   = w2g[t * 128 + tid];                              // [p*64 + i]
  if (tid < 2) s_b2[tid] = b2g[t * 2 + tid];
  __syncthreads();

  const int lane = tid & 31;
  const int wv   = tid >> 5;
  const int m0   = wv * 32;         // local row base for this wave
  const int llo  = lane & 15;
  const int lhi  = lane >> 4;       // lane-half selects K sub-range

  // ================= layer 0: h = X(16x128) @ w0[t]^T =================
  // A fragments (16x32 f16) straight from global x, ISA A-layout:
  //   lanes 0-15: K in {0..7, 16..23}; lanes 16-31: K in {8..15, 24..31}
  v16h A0[2][4];
  #pragma unroll
  for (int r = 0; r < 2; ++r) {
    #pragma unroll
    for (int kb = 0; kb < 4; ++kb) {
      const float* xr = x + (size_t)(row0 + m0 + r * 16 + llo) * 128
                          + kb * 32 + 8 * lhi;
      float4 p0 = *(const float4*)(xr);
      float4 p1 = *(const float4*)(xr + 4);
      float4 p2 = *(const float4*)(xr + 16);
      float4 p3 = *(const float4*)(xr + 20);
      v16h a;
      a[0]=(_Float16)p0.x; a[1]=(_Float16)p0.y; a[2]=(_Float16)p0.z; a[3]=(_Float16)p0.w;
      a[4]=(_Float16)p1.x; a[5]=(_Float16)p1.y; a[6]=(_Float16)p1.z; a[7]=(_Float16)p1.w;
      a[8]=(_Float16)p2.x; a[9]=(_Float16)p2.y; a[10]=(_Float16)p2.z; a[11]=(_Float16)p2.w;
      a[12]=(_Float16)p3.x; a[13]=(_Float16)p3.y; a[14]=(_Float16)p3.z; a[15]=(_Float16)p3.w;
      A0[r][kb] = a;
    }
  }

  v8f C[2][4];
  #pragma unroll
  for (int r = 0; r < 2; ++r)
    #pragma unroll
    for (int nb = 0; nb < 4; ++nb) C[r][nb] = (v8f){};

  #pragma unroll
  for (int kb = 0; kb < 4; ++kb) {
    #pragma unroll
    for (int nb = 0; nb < 4; ++nb) {
      // B(32x16): lane = N = i, per lane K in {8h..8h+7, 16+8h..16+8h+7}
      int n = nb * 16 + llo;
      int base = n * 136 + kb * 32 + 8 * lhi;
      v16h bf = join16(*(const v4u*)&s_w0[base], *(const v4u*)&s_w0[base + 16]);
      C[0][nb] = wmma16(A0[0][kb], bf, C[0][nb]);
      C[1][nb] = wmma16(A0[1][kb], bf, C[1][nb]);
    }
  }

  // epilogue 0: + b0 - diag*x[:,t], leaky, -> LDS f16
  #pragma unroll
  for (int r = 0; r < 2; ++r) {
    #pragma unroll
    for (int nb = 0; nb < 4; ++nb) {
      int i = nb * 16 + llo;
      float bi = s_b0[i], di = s_diag[i];
      int mlb = m0 + r * 16 + 8 * lhi;       // C/D layout: VGPR rr -> M = mlb+rr
      v8f c = C[r][nb];
      #pragma unroll
      for (int rr = 0; rr < 8; ++rr) {
        float v = c[rr] + bi - di * s_xcol[mlb + rr];
        s_h[(mlb + rr) * 72 + i] = (_Float16)leaky(v);
      }
    }
  }

  // ================= layer 1: h2 = h(16x64) @ w1[t]^T =================
  v16h A1[2][2];
  #pragma unroll
  for (int r = 0; r < 2; ++r) {
    #pragma unroll
    for (int kb = 0; kb < 2; ++kb) {
      int m = m0 + r * 16 + llo;
      int base = m * 72 + kb * 32 + 8 * lhi;
      A1[r][kb] = join16(*(const v4u*)&s_h[base], *(const v4u*)&s_h[base + 16]);
    }
  }

  v8f C2[2][4];
  #pragma unroll
  for (int r = 0; r < 2; ++r)
    #pragma unroll
    for (int nb = 0; nb < 4; ++nb) C2[r][nb] = (v8f){};

  #pragma unroll
  for (int kb = 0; kb < 2; ++kb) {
    #pragma unroll
    for (int nb = 0; nb < 4; ++nb) {
      int n = nb * 16 + llo;
      int base = n * 72 + kb * 32 + 8 * lhi;
      v16h bf = join16(*(const v4u*)&s_w1[base], *(const v4u*)&s_w1[base + 16]);
      C2[0][nb] = wmma16(A1[0][kb], bf, C2[0][nb]);
      C2[1][nb] = wmma16(A1[1][kb], bf, C2[1][nb]);
    }
  }

  // epilogue 1: + b1, leaky, -> LDS f16 (A1 already consumed; safe to overwrite)
  #pragma unroll
  for (int r = 0; r < 2; ++r) {
    #pragma unroll
    for (int nb = 0; nb < 4; ++nb) {
      int i = nb * 16 + llo;
      float bi = s_b1[i];
      int mlb = m0 + r * 16 + 8 * lhi;
      v8f c = C2[r][nb];
      #pragma unroll
      for (int rr = 0; rr < 8; ++rr) {
        float v = c[rr] + bi;
        s_h[(mlb + rr) * 72 + i] = (_Float16)leaky(v);
      }
    }
  }

  // ================= layer 2: out = h2 @ w2[t]^T + b2 (f32 VALU) =========
  {
    int m = m0 + lane;                      // local row 0..127
    const v2h* hp = (const v2h*)&s_h[m * 72];
    float a0 = s_b2[0], a1 = s_b2[1];
    #pragma unroll
    for (int k = 0; k < 32; ++k) {
      v2h hv = hp[k];
      float h0 = (float)hv[0], h1 = (float)hv[1];
      a0 += s_w2[2 * k] * h0 + s_w2[2 * k + 1] * h1;
      a1 += s_w2[64 + 2 * k] * h0 + s_w2[64 + 2 * k + 1] * h1;
    }
    float2 o; o.x = a0; o.y = a1;
    ((float2*)out)[(size_t)(row0 + m) * 128 + t] = o;
  }
}

extern "C" void kernel_launch(void* const* d_in, const int* in_sizes, int n_in,
                              void* d_out, int out_size, void* d_ws, size_t ws_size,
                              hipStream_t stream) {
  const float* x  = (const float*)d_in[0];
  const float* w0 = (const float*)d_in[1];
  const float* b0 = (const float*)d_in[2];
  const float* w1 = (const float*)d_in[3];
  const float* b1 = (const float*)d_in[4];
  const float* w2 = (const float*)d_in[5];
  const float* b2 = (const float*)d_in[6];
  float* out = (float*)d_out;

  const int B = in_sizes[0] / 128;   // 8192
  dim3 grid(B / 128, 128, 1);
  dim3 block(128, 1, 1);
  causal_mlp_kernel<<<grid, block, 0, stream>>>(x, w0, b0, w1, b1, w2, b2, out);
}
